// PCNet_8907762171991
// MI455X (gfx1250) — matrix operand order repaired and checked by reading, not compiled, not run on hardware
//
#include <hip/hip_runtime.h>
#include <hip/hip_bf16.h>

typedef __attribute__((ext_vector_type(16))) _Float16 v16h;
typedef __attribute__((ext_vector_type(8)))  _Float16 v8h;
typedef __attribute__((ext_vector_type(8)))  float    v8f;
typedef __attribute__((ext_vector_type(4)))  float    v4f;

#define DEPTH  10
#define HID    48
#define INDIM  784
#define OUTD   10
#define STEPS  20
#define LRATE  0.1f
#define BT     128      // batch rows per workgroup (8 waves x 16 rows)
#define BLOCK  256
#define NT     3        // 3 n-tiles of 16 cover HID=48

__device__ __forceinline__ v8f wmma16(v16h a, v16h b, v8f c) {
  return __builtin_amdgcn_wmma_f32_16x16x32_f16(false, a, false, b, (short)0, c,
                                                false, false);
}

// A operand (16x32 f16 tile) from row-major LDS, rows [rowBase,rowBase+16),
// K elements [k0,k0+32). width = valid columns per row; k >= width reads as 0.
__device__ __forceinline__ v16h ldA(const _Float16* base, int width, int rowBase,
                                    int l16, int hi, int k0) {
  const _Float16* row = base + (rowBase + l16) * width;
  int ka = k0 + hi * 8;
  int kb = k0 + 16 + hi * 8;
  v16h a;
  v8h lo = *(const v8h*)(row + ka);
#pragma unroll
  for (int j = 0; j < 8; ++j) a[j] = lo[j];
  if (kb + 8 <= width) {
    v8h up = *(const v8h*)(row + kb);
#pragma unroll
    for (int j = 0; j < 8; ++j) a[8 + j] = up[j];
  } else {
#pragma unroll
    for (int j = 0; j < 8; ++j) a[8 + j] = (_Float16)0.0f;
  }
  return a;
}

// B operand (32x16 f16 tile): lane holds column n = nt*16+l16, K from row n of
// a row-major (stride) LDS weight image. Out-of-range K is clamped; safe since
// the matching A elements are zero.
__device__ __forceinline__ v16h ldB(const _Float16* base, int stride, int kmax,
                                    int nt, int l16, int hi, int k0) {
  const _Float16* row = base + (nt * 16 + l16) * stride;
  int ka = k0 + hi * 8;
  int kb = k0 + 16 + hi * 8;
  if (kb + 8 > kmax) kb = kmax - 8;
  v8h b0 = *(const v8h*)(row + ka);
  v8h b1 = *(const v8h*)(row + kb);
  v16h b;
#pragma unroll
  for (int j = 0; j < 8; ++j) { b[j] = b0[j]; b[8 + j] = b1[j]; }
  return b;
}

// Branch-free A tile from global f32 x row (full 32-wide chunk, k0+32 <= 768)
__device__ __forceinline__ v16h ldAxF(const float* row, int hi, int k0) {
  int ka = k0 + hi * 8;
  int kb = k0 + 16 + hi * 8;
  v16h a;
#pragma unroll
  for (int j = 0; j < 8; ++j) a[j] = (_Float16)row[ka + j];
#pragma unroll
  for (int j = 0; j < 8; ++j) a[8 + j] = (_Float16)row[kb + j];
  return a;
}

// Tail A tile at k0=768 (16 valid columns, rest zero) -- constant-folds
__device__ __forceinline__ v16h ldAxT(const float* row, int hi) {
  int ka = 768 + hi * 8;
  v16h a;
#pragma unroll
  for (int j = 0; j < 8; ++j) a[j] = (_Float16)row[ka + j];
#pragma unroll
  for (int j = 0; j < 8; ++j) a[8 + j] = (_Float16)0.0f;
  return a;
}

// Branch-free B tile from global f32 W0 (full chunk)
__device__ __forceinline__ v16h ldBw0F(const float* W0, int nt, int l16, int hi,
                                       int k0) {
  int n = nt * 16 + l16;
  int ks = k0 + hi * 8;
  const float* row = W0 + n * INDIM + ks;
  v16h b;
#pragma unroll
  for (int j = 0; j < 16; ++j) b[j] = (_Float16)row[j + (j >= 8 ? 8 : 0)];
  return b;
}

// Tail B tile at k0=768: hi=0 holds k 768..775, hi=1 k 776..783; elements 8..15 repeat them (A is zero there)
__device__ __forceinline__ v16h ldBw0T(const float* W0, int nt, int l16, int hi) {
  int n = nt * 16 + l16;
  int ks = 768 + hi * 8;   // elements 0..7: k = 768+8*hi+j; elements 8..15 (A zero) re-read them
  const float* row = W0 + n * INDIM + ks;
  v16h b;
#pragma unroll
  for (int j = 0; j < 16; ++j) b[j] = (_Float16)row[j & 7];
  return b;
}

// C/D-layout elementwise access of a row-major f16 LDS image
__device__ __forceinline__ v8f ldD(const _Float16* base, int width, int rowBase,
                                   int nt, int l16, int hi) {
  int col = nt * 16 + l16;
  v8f c;
#pragma unroll
  for (int r = 0; r < 8; ++r)
    c[r] = (float)base[(rowBase + r + 8 * hi) * width + col];
  return c;
}

__device__ __forceinline__ void stD(_Float16* base, int width, int rowBase,
                                    int nt, int l16, int hi, v8f v) {
  int col = nt * 16 + l16;
#pragma unroll
  for (int r = 0; r < 8; ++r)
    base[(rowBase + r + 8 * hi) * width + col] = (_Float16)v[r];
}

// pre = v @ W^T + b for one hidden weight (48x48), this wave's 16 rows
__device__ __forceinline__ void fwd48(const _Float16* vbase, const _Float16* wbase,
                                      const float* bias, int r0, int l16, int hi,
                                      v8f p[NT]) {
  v16h a0 = ldA(vbase, HID, r0, l16, hi, 0);
  v16h a1 = ldA(vbase, HID, r0, l16, hi, 32);
#pragma unroll
  for (int t = 0; t < NT; ++t) {
    float bv = bias[t * 16 + l16];
    v8f c;
#pragma unroll
    for (int r = 0; r < 8; ++r) c[r] = bv;
    c = wmma16(a0, ldB(wbase, HID, HID, t, l16, hi, 0), c);
    c = wmma16(a1, ldB(wbase, HID, HID, t, l16, hi, 32), c);
    p[t] = c;
  }
}

__global__ __launch_bounds__(BLOCK) void pcnet_settle(
    const float* __restrict__ x, const int* __restrict__ target,
    const float* __restrict__ W0, const float* __restrict__ b0,
    const float* __restrict__ Wh, const float* __restrict__ bh,
    const float* __restrict__ Wout, const float* __restrict__ bout,
    float* __restrict__ out, int batch) {
  // ---- LDS images ----
  __shared__ __align__(16) _Float16 sW [9][HID][HID];   // forward: row n = W[n][:]
  __shared__ __align__(16) _Float16 sWT[9][HID][HID];   // backward: row n = W[:][n]
  __shared__ __align__(16) _Float16 sWo [16][HID];      // Wout rows (pad to 16, zeros)
  __shared__ __align__(16) _Float16 sWoT[HID][32];      // WoutT (o-cols 10..31 zero)
  __shared__ __align__(16) float    sbh[9][HID];
  __shared__ __align__(16) float    sbo[16];
  __shared__ __align__(16) _Float16 sV [DEPTH][BT][HID]; // value-node state (f16)
  __shared__ __align__(16) _Float16 sR0[BT][HID];        // relu(x@W0^T+b0), constant
  __shared__ __align__(16) _Float16 sE [8][16][HID];     // per-wave masked-err scratch
  __shared__ __align__(16) _Float16 sE10[8][16][32];     // per-wave output-err scratch

  const int tid  = threadIdx.x;
  const int w    = tid >> 5;
  const int lane = tid & 31;
  const int l16  = lane & 15;
  const int hi   = (lane >> 4) & 1;
  const int r0   = w * 16;                 // this wave's local row base
  const int wgBase = blockIdx.x * BT;      // this workgroup's global row base

  // ---- cooperative weight staging ----
  for (int idx = tid; idx < 9 * HID * HID; idx += BLOCK) {
    int li = idx / (HID * HID);
    int rem = idx - li * HID * HID;
    int r = rem / HID, c = rem - r * HID;
    _Float16 v = (_Float16)Wh[idx];
    sW[li][r][c]  = v;
    sWT[li][c][r] = v;
  }
  for (int idx = tid; idx < 16 * HID; idx += BLOCK) {
    int r = idx / HID, c = idx - r * HID;
    sWo[r][c] = (r < OUTD) ? (_Float16)Wout[r * HID + c] : (_Float16)0.0f;
  }
  for (int idx = tid; idx < HID * 32; idx += BLOCK) {
    int n = idx / 32, o = idx - n * 32;
    sWoT[n][o] = (o < OUTD) ? (_Float16)Wout[o * HID + n] : (_Float16)0.0f;
  }
  for (int idx = tid; idx < 9 * HID; idx += BLOCK) ((float*)sbh)[idx] = bh[idx];
  for (int idx = tid; idx < 16; idx += BLOCK) sbo[idx] = (idx < OUTD) ? bout[idx] : 0.0f;
  for (int idx = tid; idx < 8 * 16 * 32; idx += BLOCK)
    ((_Float16*)sE10)[idx] = (_Float16)0.0f;
  __syncthreads();
  // From here on every wave works only on its own 16 rows: no more barriers
  // until the final write-out.

  // ---- one-hot targets in C/D layout registers ----
  float oh[8];
#pragma unroll
  for (int r = 0; r < 8; ++r) {
    int row = wgBase + r0 + r + 8 * hi;
    int tg = target[row];
    oh[r] = (l16 < OUTD && tg == l16) ? 1.0f : 0.0f;
  }

  // ---- relu0 = relu(x @ W0^T + b0): stream K=784 from global ----
  v8f rr[NT];  // relu0 kept in registers (constant across all settle steps)
  {
    v8f acc[NT];
#pragma unroll
    for (int t = 0; t < NT; ++t) {
      float bv = b0[t * 16 + l16];
#pragma unroll
      for (int r = 0; r < 8; ++r) acc[t][r] = bv;
    }
    const float* xrow = x + (size_t)(wgBase + r0 + l16) * INDIM;
#pragma unroll 1
    for (int k0 = 0; k0 + 32 <= INDIM - 16; k0 += 32) {   // 24 full chunks
      __builtin_prefetch(xrow + k0 + 128, 0, 1);          // global_prefetch
      v16h a = ldAxF(xrow, hi, k0);
#pragma unroll
      for (int t = 0; t < NT; ++t)
        acc[t] = wmma16(a, ldBw0F(W0, t, l16, hi, k0), acc[t]);
    }
    {  // tail chunk k = 768..783
      v16h a = ldAxT(xrow, hi);
#pragma unroll
      for (int t = 0; t < NT; ++t)
        acc[t] = wmma16(a, ldBw0T(W0, t, l16, hi), acc[t]);
    }
#pragma unroll
    for (int t = 0; t < NT; ++t) {
#pragma unroll
      for (int r = 0; r < 8; ++r) rr[t][r] = fmaxf(acc[t][r], 0.0f);
      stD(&sR0[0][0],   HID, r0, t, l16, hi, rr[t]);
      stD(&sV[0][0][0], HID, r0, t, l16, hi, rr[t]);   // v_1 init
    }
  }

  // ---- init forward through hidden layers ----
  for (int i = 1; i <= 9; ++i) {
    v8f p[NT];
    fwd48(&sV[i - 1][0][0], &sW[i - 1][0][0], sbh[i - 1], r0, l16, hi, p);
#pragma unroll
    for (int t = 0; t < NT; ++t) {
      v8f rv;
#pragma unroll
      for (int r = 0; r < 8; ++r) rv[r] = fmaxf(p[t][r], 0.0f);
      stD(&sV[i][0][0], HID, r0, t, l16, hi, rv);
    }
  }

  // ---- settle loop ----
#pragma unroll 1
  for (int s = 0; s < STEPS; ++s) {
    v8f ep[NT];  // err_{i-1}, starts as err_0 = v_1 - relu0
    v8f vo[NT];  // old value of the layer about to be updated (v_i at iter i)
#pragma unroll
    for (int t = 0; t < NT; ++t) {
      vo[t] = ldD(&sV[0][0][0], HID, r0, t, l16, hi);   // v_1
#pragma unroll
      for (int r = 0; r < 8; ++r) ep[t][r] = vo[t][r] - rr[t][r];
    }

#pragma unroll 1
    for (int i = 1; i <= 9; ++i) {
      v8f p[NT];
      fwd48(&sV[i - 1][0][0], &sW[i - 1][0][0], sbh[i - 1], r0, l16, hi, p);

      v8f e[NT], vn[NT];  // err_i = v_{i+1} - relu(pre_i); masked -> sE scratch
#pragma unroll
      for (int t = 0; t < NT; ++t) {
        vn[t] = ldD(&sV[i][0][0], HID, r0, t, l16, hi);  // old v_{i+1}
#pragma unroll
        for (int r = 0; r < 8; ++r) {
          e[t][r] = vn[t][r] - fmaxf(p[t][r], 0.0f);
          sE[w][r + 8 * hi][t * 16 + l16] =
              (p[t][r] > 0.0f) ? (_Float16)e[t][r] : (_Float16)0.0f;
        }
      }

      // td_i = (err_i * (pre_i>0)) @ W  (reuse p for td)
      {
        v16h a0 = ldA(&sE[w][0][0], HID, 0, l16, hi, 0);
        v16h a1 = ldA(&sE[w][0][0], HID, 0, l16, hi, 32);
#pragma unroll
        for (int t = 0; t < NT; ++t) {
          v8f c = {};
          c = wmma16(a0, ldB(&sWT[i - 1][0][0], HID, HID, t, l16, hi, 0), c);
          c = wmma16(a1, ldB(&sWT[i - 1][0][0], HID, HID, t, l16, hi, 32), c);
          p[t] = c;
        }
      }

      // v_i <- clip(v_i - LR*(err_{i-1} - td_i));  vo carries old v_{i+1} on
#pragma unroll
      for (int t = 0; t < NT; ++t) {
        v8f nv;
#pragma unroll
        for (int r = 0; r < 8; ++r) {
          float u = vo[t][r] - LRATE * (ep[t][r] - p[t][r]);
          nv[r] = fminf(fmaxf(u, -10.0f), 10.0f);
        }
        stD(&sV[i - 1][0][0], HID, r0, t, l16, hi, nv);
        vo[t] = vn[t];
        ep[t] = e[t];
      }
    }

    // ---- output layer (i = 10): vo == old v_10 ----
    {
      v16h a0 = ldA(&sV[9][0][0], HID, r0, l16, hi, 0);
      v16h a1 = ldA(&sV[9][0][0], HID, r0, l16, hi, 32);
      float bv = sbo[l16];
      v8f po;
#pragma unroll
      for (int r = 0; r < 8; ++r) po[r] = bv;
      po = wmma16(a0, ldB(&sWo[0][0], HID, HID, 0, l16, hi, 0), po);
      po = wmma16(a1, ldB(&sWo[0][0], HID, HID, 0, l16, hi, 32), po);

      v8f e10;
#pragma unroll
      for (int r = 0; r < 8; ++r) {
        e10[r] = oh[r] - po[r];           // zero in padded cols by construction
        sE10[w][r + 8 * hi][l16] = (_Float16)e10[r];
      }

      v16h ea = ldA(&sE10[w][0][0], 32, 0, l16, hi, 0);
      v8f td[NT];
#pragma unroll
      for (int t = 0; t < NT; ++t) {
        v8f c = {};
        c = wmma16(ea, ldB(&sWoT[0][0], 32, 32, t, l16, hi, 0), c);
        td[t] = c;
      }
#pragma unroll
      for (int t = 0; t < NT; ++t) {
        v8f nv;
#pragma unroll
        for (int r = 0; r < 8; ++r) {
          float u = vo[t][r] - LRATE * (ep[t][r] - td[t][r]);
          nv[r] = fminf(fmaxf(u, -10.0f), 10.0f);
        }
        stD(&sV[9][0][0], HID, r0, t, l16, hi, nv);
      }
    }
  }

  // ---- write settled V (DEPTH, BATCH, HID) f32, vectorized float4 ----
  __syncthreads();
  for (int idx = tid; idx < DEPTH * BT * HID / 4; idx += BLOCK) {
    int layer = idx / (BT * HID / 4);
    int rem = idx - layer * (BT * HID / 4);      // float4 index inside the tile
    const _Float16* src = &sV[layer][0][0] + rem * 4;
    v4f val;
#pragma unroll
    for (int j = 0; j < 4; ++j) val[j] = (float)src[j];
    *(v4f*)(out + (size_t)layer * batch * HID + (size_t)wgBase * HID + rem * 4) =
        val;
  }
}

extern "C" void kernel_launch(void* const* d_in, const int* in_sizes, int n_in,
                              void* d_out, int out_size, void* d_ws, size_t ws_size,
                              hipStream_t stream) {
  const float* x    = (const float*)d_in[0];
  const int*   tgt  = (const int*)d_in[1];
  const float* W0   = (const float*)d_in[2];
  const float* b0   = (const float*)d_in[3];
  const float* Wh   = (const float*)d_in[4];
  const float* bh   = (const float*)d_in[5];
  const float* Wout = (const float*)d_in[6];
  const float* bout = (const float*)d_in[7];
  int batch = in_sizes[1];          // 65536
  int grid = batch / BT;            // 512 workgroups
  pcnet_settle<<<grid, BLOCK, 0, stream>>>(x, tgt, W0, b0, Wh, bh, Wout, bout,
                                           (float*)d_out, batch);
}
